// NoisyGatingRouter_23914377904858
// MI455X (gfx1250) — matrix operand order repaired and checked by reading, not compiled, not run on hardware
//
#include <hip/hip_runtime.h>
#include <hip/hip_bf16.h>
#include <math.h>

// ---------------------------------------------------------------------------
// NoisyGatingRouter for MI455X (gfx1250, wave32, WMMA)
//
// Roofline: 533 MB mandatory HBM traffic -> ~23us @ 23.3 TB/s (memory bound).
// Main GEMM (51.5 GFLOP) in bf16 WMMA 16x16x32 w/ f32 accumulation ~10us.
// 32 batch rows per block: each 1KB weight B-fragment feeds 2 WMMAs,
// halving L2->WGP weight traffic (3.0 GB -> 1.5 GB) vs 16-row blocks.
// LDS: A-tile (49.7 KB) overlaid with gating/logits buffers (dead after
// the K-loop) to stay within the static-LDS budget.
// ---------------------------------------------------------------------------

typedef __bf16  v16bf __attribute__((ext_vector_type(16)));
typedef float   v8f   __attribute__((ext_vector_type(8)));

namespace {
constexpr int B_TOT = 131072;
constexpr int D     = 256;
constexpr int K3    = 768;   // 3*D
constexpr int P     = 16;
constexpr int KR    = 4;

constexpr int M_BLK = 32;         // batch rows per block (2 m-tiles)
constexpr int A_STRIDE = K3 + 8;  // bf16 elems; pad to dodge LDS bank conflicts
constexpr int G_STRIDE = D + 8;

constexpr int A_BYTES = M_BLK * A_STRIDE * 2;           // 49664
constexpr int G_BYTES = M_BLK * G_STRIDE * 2;           // 16896
constexpr int L_OFF   = G_BYTES;                        // float[32][16]
constexpr int N_OFF   = G_BYTES + M_BLK * P * 4;        // float[32][16]

struct U32x8 { uint4 lo, hi; };

__device__ __forceinline__ v16bf load_bf16x16(const __bf16* p) {
    U32x8 u;
    u.lo = *(const uint4*)(p);
    u.hi = *(const uint4*)(p + 8);
    return __builtin_bit_cast(v16bf, u);
}

__device__ __forceinline__ float softplusf(float x) {
    // log1p(exp(x)), numerically stable (matches jax.nn.softplus)
    return (x > 0.0f) ? (x + log1pf(expf(-x))) : log1pf(expf(x));
}
} // namespace

// --- weight conversion: f32 -> bf16 (RNE) ----------------------------------
__global__ void cvt_f32_to_bf16(const float* __restrict__ src,
                                __bf16* __restrict__ dst, int n) {
    int i = blockIdx.x * blockDim.x + threadIdx.x;
    if (i < n) dst[i] = (__bf16)src[i];
}

// --- fused router ----------------------------------------------------------
__global__ __launch_bounds__(128)
void noisy_gating_router(const float* __restrict__ z_n,
                         const float* __restrict__ z_sea,
                         const float* __restrict__ z_trend,
                         const float* __restrict__ noise_z,
                         const float* __restrict__ noise_r,
                         const int*   __restrict__ patch_cand,
                         const float* __restrict__ b_proj,
                         const float* __restrict__ b_route,
                         const float* __restrict__ b_noise,
                         const __bf16* __restrict__ Wp,   // [256 x 768] bf16
                         const __bf16* __restrict__ Wr,   // [16  x 256] bf16
                         const __bf16* __restrict__ Wn,   // [16  x 256] bf16
                         float* __restrict__ outS,        // [B,16] sparse routing
                         int*   __restrict__ outP,        // [B,4] selected patches
                         int*   __restrict__ outI)        // [B,4] topk indices
{
    // Overlaid LDS: [A-tile] while K-loop runs, then [gating | logits | noise].
    __shared__ char smem[A_BYTES] __attribute__((aligned(16)));
    __bf16* A_lds = (__bf16*)smem;
    __bf16* G_lds = (__bf16*)smem;
    float*  L_lds = (float*)(smem + L_OFF);
    float*  N_lds = (float*)(smem + N_OFF);

    const int tid  = threadIdx.x;
    const int rowg = blockIdx.x * M_BLK;      // first batch row of this block

    // ---- Phase 0: stage 32x768 concat tile into LDS as bf16 ----
    for (int i = tid; i < M_BLK * (K3 / 4); i += 128) {
        const int row = i / (K3 / 4);
        const int col = (i % (K3 / 4)) * 4;
        const float* src; int cc;
        if (col < 256)      { src = z_n;     cc = col;       }
        else if (col < 512) { src = z_sea;   cc = col - 256; }
        else                { src = z_trend; cc = col - 512; }
        const float4 v = *(const float4*)(src + (size_t)(rowg + row) * D + cc);
        __bf16* d = &A_lds[row * A_STRIDE + col];
        d[0] = (__bf16)v.x; d[1] = (__bf16)v.y;
        d[2] = (__bf16)v.z; d[3] = (__bf16)v.w;
    }
    __syncthreads();

    const int lane = tid & 31;
    const int wave = tid >> 5;
    const int l16  = lane & 15;
    const int mh   = lane >> 4;   // half-wave selector (0/1)

    // ---- Phase 1: z_fused = z_concat @ W_proj.T ----
    // Wave owns 32 rows x 64 cols = 2 m-tiles x 4 n-tiles; B reused over m.
    v8f acc[2][4] = {};
    for (int k0 = 0; k0 < K3; k0 += 32) {
        // A fragments (contiguous K): lane<16 -> K k0..k0+15,
        // lane>=16 -> K k0+16..k0+31.
        v16bf a[2];
#pragma unroll
        for (int mt = 0; mt < 2; ++mt) {
            const __bf16* ap =
                &A_lds[(mt * 16 + l16) * A_STRIDE + k0 + mh * 16];
            U32x8 au;
            au.lo = *(const uint4*)(ap);
            au.hi = *(const uint4*)(ap + 8);
            a[mt] = __builtin_bit_cast(v16bf, au);
        }
#pragma unroll
        for (int t = 0; t < 4; ++t) {
            // B fragment (contiguous K, same order as A): lane holds column
            // n = tile*16 + l16, 16 consecutive K values (mh picks half).
            const int n = wave * 64 + t * 16 + l16;
            const v16bf b = load_bf16x16(Wp + (size_t)n * K3 + k0 + mh * 16);
#pragma unroll
            for (int mt = 0; mt < 2; ++mt)
                acc[mt][t] = __builtin_amdgcn_wmma_f32_16x16x32_bf16(
                    false, a[mt], false, b, (short)0, acc[mt][t], false, false);
        }
    }

    // A-tile is dead once every wave finishes its K-loop; reuse its LDS.
    __syncthreads();

    // ---- Phase 1 epilogue: + b_proj + 0.1*noise_z -> gating (bf16, LDS) ----
#pragma unroll
    for (int t = 0; t < 4; ++t) {
        const int ncol = wave * 64 + t * 16 + l16;
        const float bb = b_proj[ncol];
#pragma unroll
        for (int mt = 0; mt < 2; ++mt) {
#pragma unroll
            for (int j = 0; j < 8; ++j) {
                const int m = mt * 16 + j + mh * 8;  // D-layout row
                const float g = acc[mt][t][j] + bb +
                    0.1f * noise_z[(size_t)(rowg + m) * D + ncol];
                G_lds[m * G_STRIDE + ncol] = (__bf16)g;
            }
        }
    }
    __syncthreads();

    // ---- Phase 2 (waves 0,1): routing + noise heads for one 16-row group --
    if (wave < 2) {
        const int m0 = wave * 16;
        v8f accL = v8f{0}, accN = v8f{0};
        for (int k0 = 0; k0 < D; k0 += 32) {
            const __bf16* gp =
                &G_lds[(m0 + l16) * G_STRIDE + k0 + mh * 16];
            U32x8 au;
            au.lo = *(const uint4*)(gp);
            au.hi = *(const uint4*)(gp + 8);
            const v16bf a = __builtin_bit_cast(v16bf, au);
            const v16bf br = load_bf16x16(Wr + (size_t)l16 * D + k0 + mh * 16);
            const v16bf bn = load_bf16x16(Wn + (size_t)l16 * D + k0 + mh * 16);
            accL = __builtin_amdgcn_wmma_f32_16x16x32_bf16(
                false, a, false, br, (short)0, accL, false, false);
            accN = __builtin_amdgcn_wmma_f32_16x16x32_bf16(
                false, a, false, bn, (short)0, accN, false, false);
        }
#pragma unroll
        for (int j = 0; j < 8; ++j) {
            const int m = m0 + j + mh * 8;
            L_lds[m * P + l16] = accL[j];
            N_lds[m * P + l16] = accN[j];
        }
        // Same-wave cross-lane LDS RAW: wait for DS stores before reads.
        asm volatile("s_wait_dscnt 0x0" ::: "memory");

        if (lane < 16) {
            const int m = m0 + lane;
            const size_t r = (size_t)(rowg + m);
            float w[16];
            float mx = -INFINITY;
#pragma unroll
            for (int p = 0; p < 16; ++p) {
                const float ns = softplusf(N_lds[m * P + p] + b_noise[p]);
                const float v  = L_lds[m * P + p] + b_route[p] +
                                 noise_r[r * P + p] * ns;
                w[p] = v;
                mx = fmaxf(mx, v);
            }
            float s = 0.0f;
#pragma unroll
            for (int p = 0; p < 16; ++p) { w[p] = expf(w[p] - mx); s += w[p]; }
            const float inv = 1.0f / s;
#pragma unroll
            for (int p = 0; p < 16; ++p) w[p] *= inv;

            // top-4 by selection; strict '>' keeps lowest index on ties
            int idxs[KR];
            unsigned usedMask = 0;
#pragma unroll
            for (int kk = 0; kk < KR; ++kk) {
                float best = -1.0f; int bi = 0;
#pragma unroll
                for (int p = 0; p < 16; ++p) {
                    if (!((usedMask >> p) & 1u) && w[p] > best) {
                        best = w[p]; bi = p;
                    }
                }
                idxs[kk] = bi;
                usedMask |= (1u << bi);
            }
#pragma unroll
            for (int p = 0; p < 16; ++p)
                outS[r * P + p] = ((usedMask >> p) & 1u) ? w[p] : 0.0f;
#pragma unroll
            for (int kk = 0; kk < KR; ++kk) {
                outI[r * KR + kk] = idxs[kk];
                outP[r * KR + kk] = patch_cand[idxs[kk]];
            }
        }
    }
}

// ---------------------------------------------------------------------------
extern "C" void kernel_launch(void* const* d_in, const int* in_sizes, int n_in,
                              void* d_out, int out_size, void* d_ws, size_t ws_size,
                              hipStream_t stream) {
    const float* z_n     = (const float*)d_in[0];
    const float* z_sea   = (const float*)d_in[1];
    const float* z_trend = (const float*)d_in[2];
    const float* noise_z = (const float*)d_in[3];
    const float* noise_r = (const float*)d_in[4];
    const int*   patch   = (const int*)  d_in[5];
    const float* W_proj  = (const float*)d_in[6];   // [256 x 768]
    const float* b_proj  = (const float*)d_in[7];
    const float* W_route = (const float*)d_in[8];   // [16 x 256]
    const float* b_route = (const float*)d_in[9];
    const float* W_noise = (const float*)d_in[10];  // [16 x 256]
    const float* b_noise = (const float*)d_in[11];

    // workspace: bf16 weights
    char* ws = (char*)d_ws;
    __bf16* Wp = (__bf16*)(ws);                    // 256*768*2 = 393216 B
    __bf16* Wr = (__bf16*)(ws + 393216);           // 16*256*2  =   8192 B
    __bf16* Wn = (__bf16*)(ws + 401408);

    cvt_f32_to_bf16<<<(256 * 768 + 255) / 256, 256, 0, stream>>>(W_proj, Wp, 256 * 768);
    cvt_f32_to_bf16<<<(16 * 256 + 255) / 256, 256, 0, stream>>>(W_route, Wr, 16 * 256);
    cvt_f32_to_bf16<<<(16 * 256 + 255) / 256, 256, 0, stream>>>(W_noise, Wn, 16 * 256);

    // outputs: [B,16] f32 sparse ++ [B,4] i32 patches ++ [B,4] i32 indices
    float* outS = (float*)d_out;
    int*   outP = (int*)((char*)d_out + (size_t)B_TOT * 16 * 4);
    int*   outI = (int*)((char*)d_out + (size_t)B_TOT * 16 * 4 + (size_t)B_TOT * 4 * 4);

    noisy_gating_router<<<B_TOT / M_BLK, 128, 0, stream>>>(
        z_n, z_sea, z_trend, noise_z, noise_r, patch,
        b_proj, b_route, b_noise, Wp, Wr, Wn, outS, outP, outI);
}
